// GAT_7086696039040
// MI455X (gfx1250) — matrix operand order repaired and checked by reading, not compiled, not run on hardware
//
#include <hip/hip_runtime.h>
#include <hip/hip_bf16.h>

typedef __attribute__((ext_vector_type(16))) _Float16 v16h;
typedef __attribute__((ext_vector_type(8)))  _Float16 v8h;
typedef __attribute__((ext_vector_type(8)))  float    v8f;
typedef __attribute__((ext_vector_type(4)))  unsigned int u32x4;
typedef __attribute__((ext_vector_type(8)))  int      i32x8;
typedef __attribute__((ext_vector_type(4)))  int      i32x4;

#define SHUF16(lo, hi) __builtin_shufflevector((lo), (hi), 0,1,2,3,4,5,6,7,8,9,10,11,12,13,14,15)

__device__ __forceinline__ float lrelu_f(float v) { return v > 0.f ? v : 0.2f * v; }

// Load 8 consecutive f32 and convert to 8 packed f16 (two b128 global loads).
__device__ __forceinline__ v8h cvt8(const float* __restrict__ p) {
    const float4 f0 = *(const float4*)p;
    const float4 f1 = *(const float4*)(p + 4);
    v8h r;
    r[0] = (_Float16)f0.x; r[1] = (_Float16)f0.y; r[2] = (_Float16)f0.z; r[3] = (_Float16)f0.w;
    r[4] = (_Float16)f1.x; r[5] = (_Float16)f1.y; r[6] = (_Float16)f1.z; r[7] = (_Float16)f1.w;
    return r;
}

// ---------------------------------------------------------------------------
// WMMA GEMM: out[M,O] = act(A[M,K] @ W[O,K]^T + bias)
// Each wave owns a 16x64 output strip (1 M-tile x 4 N-tiles): the A-fragment
// is loaded once per K-step and reused by 4 WMMAs. K % 32 == 0, M % 16 == 0.
// B-fragment per lane = K {kb..kb+7, kb+16..kb+23} of one W row => no transpose.
// ---------------------------------------------------------------------------
__global__ __launch_bounds__(128) void gemm_wmma_kernel(
    const float* __restrict__ A, const float* __restrict__ W,
    const float* __restrict__ bias, float* __restrict__ out,
    int M, int K, int O, int act) {
    const int lane    = threadIdx.x & 31;
    const int tiles_o = (O + 63) >> 6;          // 64-wide super-tiles
    const int tiles   = (M >> 4) * tiles_o;
    const int tile    = blockIdx.x * 4 + (threadIdx.x >> 5);
    if (tile >= tiles) return;                  // wave-uniform: EXEC stays all-1s
    const int tm = tile / tiles_o;
    const int to = tile % tiles_o;

    const int khalf = lane >> 4;                // 0: K lo-half lanes, 1: hi-half
    const int m     = tm * 16 + (lane & 15);
    const float* Arow = A + (long)m * K + khalf * 8;   // A frag: K {kb..kb+7, kb+16..kb+23}

    const float* Wrow[4];
#pragma unroll
    for (int nt = 0; nt < 4; ++nt) {
        int o = to * 64 + nt * 16 + (lane & 15);
        if (o > O - 1) o = O - 1;               // clamp loads for ragged O
        Wrow[nt] = W + (long)o * K + khalf * 8; // B frag: K {kb..kb+7, kb+16..kb+23}
    }

    v8f acc[4] = {};
    for (int k0 = 0; k0 < K; k0 += 32) {
        if (k0 + 32 < K) __builtin_prefetch(Arow + k0 + 32, 0, 3);  // global_prefetch_b8
        const v16h a = SHUF16(cvt8(Arow + k0), cvt8(Arow + k0 + 16));
#pragma unroll
        for (int nt = 0; nt < 4; ++nt) {
            const v16h b = SHUF16(cvt8(Wrow[nt] + k0), cvt8(Wrow[nt] + k0 + 16));
            acc[nt] = __builtin_amdgcn_wmma_f32_16x16x32_f16(
                false, a, false, b, (short)0, acc[nt], false, false);
        }
    }

    const int rbase = tm * 16 + khalf * 8;      // C layout: VGPR r -> row r (+8 hi lanes)
#pragma unroll
    for (int nt = 0; nt < 4; ++nt) {
        const int col = to * 64 + nt * 16 + (lane & 15);
        if (col < O) {
            const float bv = bias ? bias[col] : 0.f;
#pragma unroll
            for (int r = 0; r < 8; ++r) {
                float v = acc[nt][r] + bv;
                if (act) v = lrelu_f(v);
                out[(long)(rbase + r) * O + col] = v;
            }
        }
    }
}

// ---------------------------------------------------------------------------
// Attention scores: a_s[n,h] = <h[n,h,:], att_src[h,:]>, same for a_d.
// ---------------------------------------------------------------------------
__global__ void scores_kernel(const float* __restrict__ h,
                              const float* __restrict__ att_src,
                              const float* __restrict__ att_dst,
                              float* __restrict__ a_s, float* __restrict__ a_d,
                              int N) {
    const int idx = blockIdx.x * blockDim.x + threadIdx.x;  // n*2 + head
    if (idx >= N * 2) return;
    const int n = idx >> 1, hd = idx & 1;
    const float* hp = h + (long)n * 128 + hd * 64;
    const float* sv = att_src + hd * 64;
    const float* dv = att_dst + hd * 64;
    float s = 0.f, d = 0.f;
#pragma unroll 8
    for (int c = 0; c < 64; ++c) { float v = hp[c]; s += v * sv[c]; d += v * dv[c]; }
    a_s[idx] = s;
    a_d[idx] = d;
}

// ---------------------------------------------------------------------------
// Fused dense-graph GAT aggregation: one block per (graph, head).
//   P[i,j] = softmax_j( lrelu(a_s[j] + a_d[i]) )   (128x128, f16 in LDS)
//   out[g*128+i, hd*64+c] = sum_j P[i,j] * V[j,c] + bias[col]
// The V tile ([128 rows x 64 f32 cols], row stride 128) is DMA'd into LDS by
// the Tensor Data Mover, then transposed/converted to the f16 B-fragment
// layout. The f32 staging tile shares LDS with the attention matrix (the
// staging copy dies, with a barrier, before P is written).
// ---------------------------------------------------------------------------
__global__ __launch_bounds__(256) void attn_aggregate_kernel(
    const float* __restrict__ h, const float* __restrict__ a_s,
    const float* __restrict__ a_d, const float* __restrict__ bias,
    float* __restrict__ out) {
    __shared__ float    sA[128];
    __shared__ float    sD[128];
    __shared__ _Float16 VT[64][128];          // V^T, f16 -> contiguous B-frags
    __shared__ union {
        float    vf[128][64];                 // TDM staging tile (phase 1)
        _Float16 pm[128][128];                // attention matrix (phase 2)
    } U;

    const int g    = blockIdx.x >> 1;
    const int hd   = blockIdx.x & 1;
    const int tid  = threadIdx.x;
    const int base = g * 128;
    const float* gsrc = h + (long)base * 128 + hd * 64;  // tile origin, stride 128

#if __has_builtin(__builtin_amdgcn_tensor_load_to_lds)
    if (tid < 32) {   // wave 0 issues the DMA (TDM ignores EXEC) and waits
        const unsigned long long ga = (unsigned long long)(size_t)(const void*)gsrc;
        const unsigned int       la = (unsigned int)(size_t)(void*)&U.vf[0][0];
        u32x4 g0;
        g0[0] = 1u;                                        // count=1, user desc
        g0[1] = la;                                        // lds_addr
        g0[2] = (unsigned int)(ga & 0xffffffffull);        // global_addr[31:0]
        g0[3] = (unsigned int)((ga >> 32) & 0x01ffffffull) // global_addr[56:32]
                | 0x80000000u;                             // type=2 ("image")
        i32x8 g1;
        g1[0] = (int)(2u << 16);    // data_size = 4 bytes
        g1[1] = (int)(64u << 16);   // tensor_dim0 = 64 (bits 79:48, low half)
        g1[2] = (int)(128u << 16);  // tensor_dim1 = 128 (bits 111:80, low half)
        g1[3] = (int)(64u << 16);   // tile_dim0 = 64 (bits 127:112)
        g1[4] = 128;                // tile_dim1 = 128 (bits 143:128)
        g1[5] = 128;                // tensor_dim0_stride = 128 elems (bits 207:160)
        g1[6] = 0;
        g1[7] = 0;
        const i32x4 z4 = {0, 0, 0, 0};
#if __clang_major__ >= 23
        const i32x8 z8 = {0, 0, 0, 0, 0, 0, 0, 0};
        __builtin_amdgcn_tensor_load_to_lds(g0, g1, z4, z4, z8, 0);
#else
        __builtin_amdgcn_tensor_load_to_lds(g0, g1, z4, z4, 0);
#endif
        __builtin_amdgcn_s_wait_tensorcnt((short)0);       // s_wait_tensorcnt 0
    }
#else
    for (int idx = tid; idx < 128 * 64; idx += 256)        // fallback: plain loads
        U.vf[idx >> 6][idx & 63] = gsrc[(long)(idx >> 6) * 128 + (idx & 63)];
#endif
    if (tid < 128) {
        sA[tid] = a_s[(base + tid) * 2 + hd];
        sD[tid] = a_d[(base + tid) * 2 + hd];
    }
    __syncthreads();   // TDM tile + scores visible to the whole workgroup

    // Transpose/convert staging tile -> VT (f16). Last readers of U.vf.
    for (int idx = tid; idx < 64 * 128; idx += 256) {
        const int c = idx >> 7, j = idx & 127;
        VT[c][j] = (_Float16)U.vf[j][c];
    }
    __syncthreads();   // U.vf dead; U.pm may now be written

    if (tid < 128) {   // waves 0-3 (wave-uniform branch): row-wise softmax
        const int   i  = tid;
        const float ad = sD[i];
        float mx = -3.4e38f;
        for (int j = 0; j < 128; ++j) mx = fmaxf(mx, lrelu_f(sA[j] + ad));
        float den = 0.f;
        for (int j = 0; j < 128; ++j) {
            const float ex = __expf(lrelu_f(sA[j] + ad) - mx);
            den += ex;
            U.pm[i][j] = (_Float16)ex;
        }
        const float inv = 1.f / den;
        for (int j = 0; j < 128; ++j)
            U.pm[i][j] = (_Float16)((float)U.pm[i][j] * inv);
    }
    __syncthreads();

    const int wave  = tid >> 5;
    const int lane  = tid & 31;
    const int khalf = lane >> 4;
    const int mrow  = wave * 16 + (lane & 15);

    for (int nt = 0; nt < 4; ++nt) {
        v8f acc = {};
        const int ncol = nt * 16 + (lane & 15);
#pragma unroll
        for (int ks = 0; ks < 4; ++ks) {
            const int kb  = ks * 32 + khalf * 8;
            const int kb2 = ks * 32 + khalf * 8;
            v8h alo = *(const v8h*)&U.pm[mrow][kb];
            v8h ahi = *(const v8h*)&U.pm[mrow][kb + 16];
            v8h blo = *(const v8h*)&VT[ncol][kb2];
            v8h bhi = *(const v8h*)&VT[ncol][kb2 + 16];
            v16h a = SHUF16(alo, ahi);
            v16h b = SHUF16(blo, bhi);
            acc = __builtin_amdgcn_wmma_f32_16x16x32_f16(
                false, a, false, b, (short)0, acc, false, false);
        }
        const int   col   = hd * 64 + nt * 16 + (lane & 15);
        const float bv    = bias[col];
        const int   rbase = base + wave * 16 + khalf * 8;
#pragma unroll
        for (int r = 0; r < 8; ++r)
            out[(long)(rbase + r) * 128 + col] = acc[r] + bv;
    }
}

// ---------------------------------------------------------------------------
// BatchNorm over axis 0 of x[M, C], in place: one block per channel.
// ---------------------------------------------------------------------------
__global__ __launch_bounds__(256) void bn_kernel(float* __restrict__ x,
                                                 const float* __restrict__ g,
                                                 const float* __restrict__ b,
                                                 int M, int C) {
    const int c = blockIdx.x;
    const int t = threadIdx.x;
    __shared__ float s1[256], s2[256];
    float sum = 0.f, sq = 0.f;
    for (int r = t; r < M; r += 256) {
        const float v = x[(long)r * C + c];
        sum += v; sq += v * v;
    }
    s1[t] = sum; s2[t] = sq;
    __syncthreads();
    for (int ofs = 128; ofs > 0; ofs >>= 1) {
        if (t < ofs) { s1[t] += s1[t + ofs]; s2[t] += s2[t + ofs]; }
        __syncthreads();
    }
    const float mean  = s1[0] / (float)M;
    const float var   = s2[0] / (float)M - mean * mean;
    const float scale = rsqrtf(var + 1e-5f) * g[c];
    const float shift = b[c];
    for (int r = t; r < M; r += 256)
        x[(long)r * C + c] = (x[(long)r * C + c] - mean) * scale + shift;
}

// ---------------------------------------------------------------------------
static inline void launch_gemm(const float* A, const float* W, const float* bias,
                               float* out, int M, int K, int O, int act,
                               hipStream_t stream) {
    const int tiles  = (M >> 4) * ((O + 63) >> 6);
    const int blocks = (tiles + 3) >> 2;
    gemm_wmma_kernel<<<blocks, 128, 0, stream>>>(A, W, bias, out, M, K, O, act);
}

extern "C" void kernel_launch(void* const* d_in, const int* in_sizes, int n_in,
                              void* d_out, int out_size, void* d_ws, size_t ws_size,
                              hipStream_t stream) {
    (void)in_sizes; (void)n_in; (void)out_size; (void)ws_size;
    constexpr int N = 8192;      // B_GRAPHS * P_NODES
    constexpr int HC = 128;      // HEADS * C_HID

    const float* x        = (const float*)d_in[0];
    // d_in[1] edge_index, d_in[2] edge_attr, d_in[3] batch: dense block
    // structure is known statically; attention never uses edge_attr.
    const float* w_lin0   = (const float*)d_in[4];
    const float* att_src0 = (const float*)d_in[5];
    const float* att_dst0 = (const float*)d_in[6];
    const float* bias0    = (const float*)d_in[7];
    const float* w_post0  = (const float*)d_in[8];
    const float* b_post0  = (const float*)d_in[9];
    const float* bn0_g    = (const float*)d_in[10];
    const float* bn0_b    = (const float*)d_in[11];
    const float* w_lin1   = (const float*)d_in[12];
    const float* att_src1 = (const float*)d_in[13];
    const float* att_dst1 = (const float*)d_in[14];
    const float* bias1    = (const float*)d_in[15];
    const float* w_mid1   = (const float*)d_in[16];
    const float* b_mid1   = (const float*)d_in[17];
    const float* w_node1  = (const float*)d_in[18];
    const float* b_node1  = (const float*)d_in[19];
    const float* bn1_g    = (const float*)d_in[20];
    const float* bn1_b    = (const float*)d_in[21];
    const float* w_f1     = (const float*)d_in[22];
    const float* b_f1     = (const float*)d_in[23];
    const float* w_f2     = (const float*)d_in[24];
    const float* b_f2     = (const float*)d_in[25];
    const float* w_f3     = (const float*)d_in[26];
    const float* b_f3     = (const float*)d_in[27];
    float* outp = (float*)d_out;

    float* ws  = (float*)d_ws;
    float* h0  = ws;               // N*128
    float* as0 = h0  + N * HC;     // N*2
    float* ad0 = as0 + N * 2;      // N*2
    float* g0  = ad0 + N * 2;      // N*128
    float* z0  = g0  + N * HC;     // N*64
    float* h1  = z0  + N * 64;     // N*128
    float* as1 = h1  + N * HC;     // N*2
    float* ad1 = as1 + N * 2;      // N*2
    float* g1  = ad1 + N * 2;      // N*128
    float* zm  = g1  + N * HC;     // N*64
    float* zn  = zm  + N * 64;     // N*8
    float* f1o = zn  + N * 8;      // 64*256
    float* f2o = f1o + 64 * 256;   // 64*32

    // ---- layer 0: GAT -> Linear -> LReLU -> BN ----
    launch_gemm(x, w_lin0, nullptr, h0, N, 128, 128, 0, stream);
    scores_kernel<<<(N * 2 + 255) / 256, 256, 0, stream>>>(h0, att_src0, att_dst0,
                                                           as0, ad0, N);
    attn_aggregate_kernel<<<128, 256, 0, stream>>>(h0, as0, ad0, bias0, g0);
    launch_gemm(g0, w_post0, b_post0, z0, N, 128, 64, 1, stream);
    bn_kernel<<<64, 256, 0, stream>>>(z0, bn0_g, bn0_b, N, 64);

    // ---- layer 1: GAT -> Linear(64) -> LReLU -> Linear(8) -> LReLU -> BN ----
    launch_gemm(z0, w_lin1, nullptr, h1, N, 64, 128, 0, stream);
    scores_kernel<<<(N * 2 + 255) / 256, 256, 0, stream>>>(h1, att_src1, att_dst1,
                                                           as1, ad1, N);
    attn_aggregate_kernel<<<128, 256, 0, stream>>>(h1, as1, ad1, bias1, g1);
    launch_gemm(g1, w_mid1, b_mid1, zm, N, 128, 64, 1, stream);
    launch_gemm(zm, w_node1, b_node1, zn, N, 64, 8, 1, stream);
    bn_kernel<<<8, 256, 0, stream>>>(zn, bn1_g, bn1_b, N, 8);

    // ---- concat pooling head: zn viewed as [64, 1024] ----
    launch_gemm(zn, w_f1, b_f1, f1o, 64, 1024, 256, 1, stream);
    launch_gemm(f1o, w_f2, b_f2, f2o, 64, 256, 32, 1, stream);
    launch_gemm(f2o, w_f3, b_f3, outp, 64, 32, 1, 0, stream);
}
